// RGCN_v1_2164663517562
// MI455X (gfx1250) — matrix operand order repaired and checked by reading, not compiled, not run on hardware
//
#include <hip/hip_runtime.h>
#include <hip/hip_bf16.h>

// ---------------- problem constants ----------------
#define NN 50000
#define NE 600000
#define DD 128
#define HH 128
#define RR 8

typedef __attribute__((ext_vector_type(16))) __bf16 v16bf;
typedef __attribute__((ext_vector_type(8)))  __bf16 v8bf;
typedef __attribute__((ext_vector_type(8)))  float  v8f;

// ---------------- small utility kernels ----------------
__global__ void f2bf_kernel(__bf16* __restrict__ o, const float* __restrict__ in, int n) {
    int i = blockIdx.x * blockDim.x + threadIdx.x;
    if (i < n) o[i] = (__bf16)in[i];
}

__global__ void zero_kernel(float* __restrict__ p, int n) {
    int i = blockIdx.x * blockDim.x + threadIdx.x;
    if (i < n) p[i] = 0.0f;
}

// Build transposed bf16 weight matrix Bt[col][k], col in [0,1152): 8 relations then root.
__global__ void wtrans_kernel(__bf16* __restrict__ Bt,
                              const float* __restrict__ relw,   // [8][128][128] (k -> h)
                              const float* __restrict__ rootw)  // [128][128]
{
    int i = blockIdx.x * blockDim.x + threadIdx.x;   // i = col*128 + k
    if (i >= 1152 * 128) return;
    int col = i >> 7;
    int k   = i & 127;
    float v;
    if (col < 1024) {
        int r = col >> 7;
        int h = col & 127;
        v = relw[((size_t)r * 128 + k) * 128 + h];
    } else {
        int h = col - 1024;
        v = rootw[(size_t)k * 128 + h];
    }
    Bt[(size_t)col * 128 + k] = (__bf16)v;
}

// Transposed bf16 projection weights: Bt[col][k], col<128, k<256
__global__ void projtrans_kernel(__bf16* __restrict__ Bt, const float* __restrict__ pw) {
    int i = blockIdx.x * blockDim.x + threadIdx.x;   // i = col*256 + k
    if (i >= 128 * 256) return;
    int col = i >> 8;
    int k   = i & 255;
    Bt[(size_t)col * 256 + k] = (__bf16)pw[(size_t)k * 128 + col];
}

// cnt[dst*8 + etype] += 1  (layer-invariant)
__global__ void count_kernel(float* __restrict__ cnt, const int* __restrict__ ei,
                             const int* __restrict__ et, int E) {
    int e = blockIdx.x * blockDim.x + threadIdx.x;
    if (e >= E) return;
    int d = ei[E + e];
    int t = et[e];
    atomicAdd(&cnt[(size_t)d * RR + t], 1.0f);
}

// acc = root term (last 128 cols of xW) + bias
__global__ void accinit_kernel(float* __restrict__ acc, const float* __restrict__ xW,
                               const float* __restrict__ bias, int total) {
    int i = blockIdx.x * blockDim.x + threadIdx.x;
    if (i >= total) return;
    int v = i >> 7;
    int j = i & 127;
    acc[i] = xW[(size_t)v * 1152 + 1024 + j] + bias[j];
}

// Per-edge gather of xW[src, etype] * (1/cnt) scattered with float atomics (32 lanes/edge).
__global__ void scatter_kernel(float* __restrict__ acc, const float* __restrict__ xW,
                               const float* __restrict__ cnt, const int* __restrict__ ei,
                               const int* __restrict__ et, int E) {
    int e = blockIdx.x * 8 + (threadIdx.x >> 5);
    if (e >= E) return;
    int lane = threadIdx.x & 31;
    int s = ei[e];
    int d = ei[E + e];
    int t = et[e];
    float c = cnt[(size_t)d * RR + t];
    float w = 1.0f / (c > 1.0f ? c : 1.0f);
    const float4 m = *(const float4*)(xW + (size_t)s * 1152 + (size_t)t * 128 + lane * 4);
    float* dp = acc + (size_t)d * 128 + lane * 4;
    atomicAdd(dp + 0, m.x * w);
    atomicAdd(dp + 1, m.y * w);
    atomicAdd(dp + 2, m.z * w);
    atomicAdd(dp + 3, m.w * w);
}

// ReLU -> fp32 rep output (d_out region) and bf16 copy into concat buffer for next GEMM.
__global__ void relu_store_kernel(float* __restrict__ rep, __bf16* __restrict__ hcat,
                                  int colofs, const float* __restrict__ acc, int total) {
    int i = blockIdx.x * blockDim.x + threadIdx.x;
    if (i >= total) return;
    int v = i >> 7;
    int j = i & 127;
    float r = acc[i];
    r = r > 0.0f ? r : 0.0f;
    rep[i] = r;
    hcat[(size_t)v * 256 + colofs + j] = (__bf16)r;
}

// ---------------- bf16 WMMA GEMM ----------------
// C[m, n] = sum_k A[m,k] * B[k,n] (+ bias[n])
// A: bf16 row-major [M, lda]; Bt: bf16 TRANSPOSED [Ncols][K]; C: fp32 [M, ldc].
// Block: 256 threads = 8 waves; each wave: 16 rows x 128 cols (8 WMMA tiles).
// Whole 128-col x K B-panel staged ONCE in (padded) dynamic LDS; K-loop is barrier-free.
// Row pad of 8 bf16 -> row stride = (K+8)/2 dwords = 4 (mod 64): b128 frag reads are
// bank-conflict-free across the 16 lanes.
template <int K>
__global__ void __launch_bounds__(256)
wmma_gemm_bf16(const __bf16* __restrict__ A, int lda,
               const __bf16* __restrict__ Bt,
               float* __restrict__ C, int ldc,
               int M, const float* __restrict__ bias)
{
    extern __shared__ char smem_raw[];
    __bf16* Bs = (__bf16*)smem_raw;                  // [128][K+8]
    constexpr int LDB = K + 8;

    const int tid    = threadIdx.x;
    const int wave   = tid >> 5;
    const int lane   = tid & 31;
    const int lane15 = lane & 15;
    const int khalf  = lane >> 4;
    const int n_base = blockIdx.y * 128;
    const int m0     = (blockIdx.x * 8 + wave) * 16;
    int row  = m0 + lane15;
    int rowc = row < M ? row : (M - 1);              // clamp so EXEC stays all-1s

    // ---- stage the whole B panel once (2 threads per row, 16-bf16 chunks) ----
    {
        const int r    = tid >> 1;                   // 0..127
        const int half = tid & 1;
        const __bf16* src = Bt + (size_t)(n_base + r) * K;
        __bf16*       dst = Bs + r * LDB;
#pragma unroll
        for (int i = half * 16; i < K; i += 32) {
            *(v8bf*)(dst + i)     = *(const v8bf*)(src + i);
            *(v8bf*)(dst + i + 8) = *(const v8bf*)(src + i + 8);
        }
    }
    __syncthreads();

    v8f accv[8];
    const v8f zero = {0.f, 0.f, 0.f, 0.f, 0.f, 0.f, 0.f, 0.f};
#pragma unroll
    for (int t = 0; t < 8; ++t) accv[t] = zero;

#pragma unroll
    for (int k0 = 0; k0 < K; k0 += 32) {
        // A fragment: contiguous k = 16*khalf + s per lane (matches B)
        const __bf16* ap = A + (size_t)rowc * lda + k0;
        v16bf afrag = *(const v16bf*)(ap + khalf * 16);

        // load all 8 B fragments first, then issue the WMMA chain back-to-back
        v16bf bfr[8];
#pragma unroll
        for (int t = 0; t < 8; ++t)
            bfr[t] = *(const v16bf*)&Bs[(t * 16 + lane15) * LDB + k0 + khalf * 16];
#pragma unroll
        for (int t = 0; t < 8; ++t)
            accv[t] = __builtin_amdgcn_wmma_f32_16x16x32_bf16(
                false, afrag, false, bfr[t], (short)0, accv[t], false, false);
    }

    // epilogue: C/D layout -> row = m0 + v + 8*khalf, col = n_base + t*16 + lane15
#pragma unroll
    for (int t = 0; t < 8; ++t) {
        int col = n_base + t * 16 + lane15;
        float badd = bias ? bias[col] : 0.0f;
#pragma unroll
        for (int v = 0; v < 8; ++v) {
            int r = m0 + v + (khalf ? 8 : 0);
            if (r < M) C[(size_t)r * ldc + col] = accv[t][v] + badd;
        }
    }
}

// ---------------- launcher ----------------
extern "C" void kernel_launch(void* const* d_in, const int* in_sizes, int n_in,
                              void* d_out, int out_size, void* d_ws, size_t ws_size,
                              hipStream_t stream) {
    const float* node_feat = (const float*)d_in[0];
    const int*   ei        = (const int*)d_in[1];   // [2, E]: src then dst
    const int*   et        = (const int*)d_in[2];
    const float* rel_w     = (const float*)d_in[3]; // [2][8][128][128]
    const float* root_w    = (const float*)d_in[4]; // [2][128][128]
    const float* bias      = (const float*)d_in[5]; // [2][128]
    const float* proj_w    = (const float*)d_in[6]; // [256][128]
    const float* proj_b    = (const float*)d_in[7]; // [128]
    float* out = (float*)d_out;   // final [N,128] | rep1 [N,128] | rep2 [N,128]

    // workspace layout (all offsets 256B-aligned)
    char* w = (char*)d_ws;
    float*  xW   = (float*)(w);                               // 50000*1152*4 = 230,400,000
    float*  acc  = (float*)(w + 230400000);                   // 25,600,000
    float*  cnt  = (float*)(w + 256000000);                   // 1,600,000
    __bf16* xbf  = (__bf16*)(w + 257600000);                  // 12,800,000
    __bf16* hcat = (__bf16*)(w + 270400000);                  // 25,600,000
    __bf16* Bt   = (__bf16*)(w + 296000000);                  // <= 294,912

    const int NH = NN * HH;   // 6,400,000

    // features -> bf16 (once); edge counts are layer-invariant (once)
    f2bf_kernel<<<(NH + 255) / 256, 256, 0, stream>>>(xbf, node_feat, NH);
    zero_kernel<<<(NN * RR + 255) / 256, 256, 0, stream>>>(cnt, NN * RR);
    count_kernel<<<(NE + 255) / 256, 256, 0, stream>>>(cnt, ei, et, NE);

    const dim3 gGemm(391, 9), gProj(391, 1);
    const size_t lds128 = 128 * (128 + 8) * sizeof(__bf16);   // 34,816 B
    const size_t lds256 = 128 * (256 + 8) * sizeof(__bf16);   // 67,584 B

    for (int l = 0; l < 2; ++l) {
        wtrans_kernel<<<(1152 * 128 + 255) / 256, 256, 0, stream>>>(
            Bt, rel_w + (size_t)l * RR * DD * HH, root_w + (size_t)l * DD * HH);

        const __bf16* Ain = (l == 0) ? xbf : hcat;   // layer2 reads rep1 cols of hcat
        int lda = (l == 0) ? 128 : 256;
        wmma_gemm_bf16<128><<<gGemm, 256, lds128, stream>>>(Ain, lda, Bt, xW, 1152, NN, nullptr);

        accinit_kernel<<<(NH + 255) / 256, 256, 0, stream>>>(acc, xW, bias + l * HH, NH);
        scatter_kernel<<<(NE + 7) / 8, 256, 0, stream>>>(acc, xW, cnt, ei, et, NE);
        relu_store_kernel<<<(NH + 255) / 256, 256, 0, stream>>>(
            out + (size_t)(l + 1) * NH, hcat, l * HH, acc, NH);
    }

    // final = concat(rep1, rep2) @ proj_w + proj_b
    projtrans_kernel<<<(128 * 256 + 255) / 256, 256, 0, stream>>>(Bt, proj_w);
    wmma_gemm_bf16<256><<<gProj, 256, lds256, stream>>>(hcat, 256, Bt, out, 128, NN, proj_b);
}
